// TreeCompositionLSTM_66529043415156
// MI455X (gfx1250) — matrix operand order repaired and checked by reading, not compiled, not run on hardware
//
#include <hip/hip_runtime.h>
#include <hip/hip_bf16.h>

// ---------------- problem constants ----------------
#define T_STEPS 128
#define B_SZ    256
#define WD      300
#define TD      64
#define RD      64
#define M_SZ    512
#define G_SZ    2560          // 5*M
#define KX      960           // 300+64+64+512 = 940, padded to 32-multiple
#define KTOT    1472          // KX + M_SZ (46 * 32)
#define KITER   (KTOT / 32)   // 46
#define KHALF   (KITER / 2)   // 23 per k-split wave (odd -> 11 pipelined pairs + tail)

typedef __bf16 bf16_t;
typedef bf16_t v16bf __attribute__((ext_vector_type(16)));
typedef bf16_t v8bf  __attribute__((ext_vector_type(8)));
typedef float  v8f   __attribute__((ext_vector_type(8)));

union BV16 { v16bf v; v8bf h[2]; };

__device__ __forceinline__ float fsig(float x)  { return 1.0f / (1.0f + __expf(-x)); }
__device__ __forceinline__ float ftanh(float x) { float e = __expf(2.0f * x); return (e - 1.0f) / (e + 1.0f); }

// ---------------- prep: fused transposed bf16 weight matrix ----------------
// WallT[g][k], k in [0,1472): [Ww ; Wt ; Wr ; Wk ; zero-pad ; Wh]^T
__global__ __launch_bounds__(256)
void convert_weights(const float* __restrict__ Ww, const float* __restrict__ Wt,
                     const float* __restrict__ Wr, const float* __restrict__ Wk,
                     const float* __restrict__ Wh, bf16_t* __restrict__ WallT)
{
    size_t idx = (size_t)blockIdx.x * 256 + threadIdx.x;
    if (idx >= (size_t)G_SZ * KTOT) return;
    int gcol = (int)(idx / KTOT);
    int k    = (int)(idx % KTOT);
    float v = 0.0f;
    if      (k < WD)               v = Ww[(size_t)k * G_SZ + gcol];
    else if (k < WD+TD)            v = Wt[(size_t)(k - WD) * G_SZ + gcol];
    else if (k < WD+TD+RD)         v = Wr[(size_t)(k - WD - TD) * G_SZ + gcol];
    else if (k < WD+TD+RD+M_SZ)    v = Wk[(size_t)(k - WD - TD - RD) * G_SZ + gcol];
    else if (k < KX)               v = 0.0f;               // pad
    else                           v = Wh[(size_t)(k - KX) * G_SZ + gcol];
    WallT[idx] = (bf16_t)v;
}

// ---------------- prep: bf16 concat of per-step features ----------------
// xcat[t*B + b][k], k in [0,960): [word | tag | rel | k | pad]
__global__ __launch_bounds__(256)
void convert_inputs(const float* __restrict__ word, const float* __restrict__ tag,
                    const float* __restrict__ rel,  const float* __restrict__ kf,
                    bf16_t* __restrict__ xcat)
{
    size_t idx = (size_t)blockIdx.x * 256 + threadIdx.x;
    if (idx >= (size_t)T_STEPS * B_SZ * KX) return;
    size_t tb = idx / KX;
    int   k   = (int)(idx % KX);
    float v = 0.0f;
    if      (k < WD)            v = word[tb * WD + k];
    else if (k < WD+TD)         v = tag [tb * TD + (k - WD)];
    else if (k < WD+TD+RD)      v = rel [tb * RD + (k - WD - TD)];
    else if (k < WD+TD+RD+M_SZ) v = kf  [tb * M_SZ + (k - WD - TD - RD)];
    xcat[idx] = (bf16_t)v;
}

// ---------------- prep: initial recurrent state (into ping buffer 0) ----------------
__global__ __launch_bounds__(256)
void init_state(const float* __restrict__ h0, const float* __restrict__ c0,
                float* __restrict__ c_buf0, bf16_t* __restrict__ hbf_buf0)
{
    int idx = blockIdx.x * 256 + threadIdx.x;    // 131072 = 256*512
    c_buf0[idx]   = c0[idx];
    hbf_buf0[idx] = (bf16_t)h0[idx];
}

// ---------------- fused per-step kernel (2-way K-split, SW-pipelined) ----------------
// A PAIR of waves owns all five gate tiles of one 16x16 (batch-row, m-col) block:
//   ks=0 wave: k-iters [0,23) + bias;  ks=1 wave: k-iters [23,46).
// Inner loop is an explicit 2-stage software pipeline with two NAMED register
// buffer sets so the allocator cannot collapse them: WMMAs of iter k consume a
// buffer loaded a full iteration earlier while iter k+1's 12 b128 loads fly.
// Partials combined through LDS (ds b128 + block barrier); ks=0 wave applies
// the LSTM gates in-register. h/c ping-pong across steps (race-free).
__global__ __launch_bounds__(256)
void step_fused(const bf16_t* __restrict__ xcat, const bf16_t* __restrict__ WallT,
                const float* __restrict__ bias, const float* __restrict__ q,
                const bf16_t* __restrict__ hbf_in, const float* __restrict__ c_in,
                bf16_t* __restrict__ hbf_out, float* __restrict__ c_out,
                float* __restrict__ out, int t)
{
    __shared__ v8f redbuf[4][5][32];                    // 20 KB: per-pair partial tiles

    const int lane = threadIdx.x;
    const int hf   = lane >> 4;
    const int sub  = lane & 15;
    const int pair = threadIdx.y >> 1;                  // tile slot in block: 0..3
    const int ks   = threadIdx.y & 1;                   // k-split half: 0 or 1
    const int tile = blockIdx.x * 4 + pair;             // 0 .. 511
    const int r0   = (tile >> 5) * 16;                  // batch-row tile (16 tiles)
    const int c0   = (tile & 31) * 16;                  // m-column tile (32 tiles)
    const int mrow = r0 + sub;                          // A row this lane loads
    const int mcol = c0 + sub;                          // m column of D this lane holds

    const bf16_t* __restrict__ xrow = xcat   + ((size_t)t * B_SZ + mrow) * KX;
    const bf16_t* __restrict__ hrow = hbf_in + (size_t)mrow * M_SZ;

    const bf16_t* wrow[5];
    v8f acc[5];
#pragma unroll
    for (int j2 = 0; j2 < 5; ++j2) {
        wrow[j2] = WallT + (size_t)(j2 * M_SZ + mcol) * KTOT;
        const float bv = (ks == 0) ? bias[j2 * M_SZ + mcol] : 0.0f;
#pragma unroll
        for (int j = 0; j < 8; ++j) acc[j2][j] = bv;
    }

    // issue all 12 b128 loads of k-iteration kk into (a, bm)
    auto load_iter = [&](int kk, BV16& a, BV16 (&bm)[5]) {
        const int kbase = kk * 32;
        const bf16_t* asrc = (kbase < KX) ? (xrow + kbase) : (hrow + (kbase - KX));
        a.h[0] = *(const v8bf*)(asrc + hf * 8);
        a.h[1] = *(const v8bf*)(asrc + 16 + hf * 8);
#pragma unroll
        for (int j2 = 0; j2 < 5; ++j2) {
            const bf16_t* bsrc = wrow[j2] + kbase + hf * 8;
            bm[j2].h[0] = *(const v8bf*)(bsrc);
            bm[j2].h[1] = *(const v8bf*)(bsrc + 16);
            __builtin_prefetch(wrow[j2] + kbase + 256, 0, 1);   // 8 iters ahead
        }
    };
    // fire the 5 WMMAs (one shared A tile) of a buffered k-iteration
    auto wmma5 = [&](const BV16& a, const BV16 (&bm)[5]) {
#pragma unroll
        for (int j2 = 0; j2 < 5; ++j2)
            acc[j2] = __builtin_amdgcn_wmma_f32_16x16x32_bf16(
                          false, a.v, false, bm[j2].v, (short)0, acc[j2], false, false);
    };

    const int kk0 = ks * KHALF;
    BV16 aA, bA[5], aB, bB[5];                          // two named pipeline stages
    load_iter(kk0, aA, bA);
#pragma unroll 2
    for (int i = 0; i < (KHALF - 1) / 2; ++i) {         // 11 double-iterations
        const int kk = kk0 + 2 * i;
        load_iter(kk + 1, aB, bB);
        wmma5(aA, bA);                                  // consumes loads from iter kk
        load_iter(kk + 2, aA, bA);
        wmma5(aB, bB);                                  // consumes loads from iter kk+1
    }
    wmma5(aA, bA);                                      // drain tail (kk0 + 22)

    // ---- cross-wave reduction of the K-split partials through LDS ----
    if (ks == 1) {
#pragma unroll
        for (int j2 = 0; j2 < 5; ++j2) redbuf[pair][j2][lane] = acc[j2];
    }
    __syncthreads();
    if (ks == 1) return;                                // upper-half waves done
#pragma unroll
    for (int j2 = 0; j2 < 5; ++j2) {
        const v8f part = redbuf[pair][j2][lane];
#pragma unroll
        for (int j = 0; j < 8; ++j) acc[j2][j] += part[j];
    }

    // ---- gates in-register: acc[0]=gi acc[1]=gfd acc[2]=gfl acc[3]=go acc[4]=gu ----
    const float* __restrict__ qstep = q + (size_t)t * B_SZ * M_SZ;
#pragma unroll
    for (int j = 0; j < 8; ++j) {
        const int    rb   = r0 + hf * 8 + j;            // batch row
        const size_t sidx = (size_t)rb * M_SZ + mcol;   // state index
        const float i_ = fsig (acc[0][j]);
        const float fd = fsig (acc[1][j]);
        const float fl = fsig (acc[2][j]);
        const float o_ = fsig (acc[3][j]);
        const float u_ = ftanh(acc[4][j]);
        const float cn = i_ * u_ + fd * qstep[sidx] + fl * c_in[sidx];
        const float hn = o_ * ftanh(cn);
        c_out[sidx]   = cn;
        hbf_out[sidx] = (bf16_t)hn;
        if (out) {                                      // final step -> concat([h, c], -1)
            out[(size_t)rb * (2 * M_SZ) + mcol]        = hn;
            out[(size_t)rb * (2 * M_SZ) + M_SZ + mcol] = cn;
        }
    }
}

// ---------------- launcher ----------------
static inline size_t align256(size_t x) { return (x + 255) & ~(size_t)255; }

extern "C" void kernel_launch(void* const* d_in, const int* in_sizes, int n_in,
                              void* d_out, int out_size, void* d_ws, size_t ws_size,
                              hipStream_t stream)
{
    (void)in_sizes; (void)n_in; (void)out_size; (void)ws_size;
    const float* word = (const float*)d_in[0];
    const float* tag  = (const float*)d_in[1];
    const float* rel  = (const float*)d_in[2];
    const float* kf   = (const float*)d_in[3];
    const float* q    = (const float*)d_in[4];
    const float* h0   = (const float*)d_in[5];
    const float* c0   = (const float*)d_in[6];
    const float* Ww   = (const float*)d_in[7];
    const float* Wt   = (const float*)d_in[8];
    const float* Wr   = (const float*)d_in[9];
    const float* Wk   = (const float*)d_in[10];
    const float* Wh   = (const float*)d_in[11];
    const float* bias = (const float*)d_in[12];
    float* out = (float*)d_out;

    // workspace carve-up (~73 MB total)
    char* ws = (char*)d_ws;
    size_t off = 0;
    bf16_t* WallT  = (bf16_t*)(ws + off); off += align256((size_t)G_SZ * KTOT * sizeof(bf16_t));
    bf16_t* xcat   = (bf16_t*)(ws + off); off += align256((size_t)T_STEPS * B_SZ * KX * sizeof(bf16_t));
    float*  c_pp[2];
    bf16_t* h_pp[2];
    c_pp[0] = (float*) (ws + off); off += align256((size_t)B_SZ * M_SZ * sizeof(float));
    c_pp[1] = (float*) (ws + off); off += align256((size_t)B_SZ * M_SZ * sizeof(float));
    h_pp[0] = (bf16_t*)(ws + off); off += align256((size_t)B_SZ * M_SZ * sizeof(bf16_t));
    h_pp[1] = (bf16_t*)(ws + off); off += align256((size_t)B_SZ * M_SZ * sizeof(bf16_t));

    {   // weight transpose+convert: G_SZ*KTOT elements
        size_t n = (size_t)G_SZ * KTOT;
        convert_weights<<<(unsigned)((n + 255) / 256), 256, 0, stream>>>(Ww, Wt, Wr, Wk, Wh, WallT);
    }
    {   // input concat+convert: T*B*KX elements
        size_t n = (size_t)T_STEPS * B_SZ * KX;
        convert_inputs<<<(unsigned)((n + 255) / 256), 256, 0, stream>>>(word, tag, rel, kf, xcat);
    }
    init_state<<<(B_SZ * M_SZ) / 256, 256, 0, stream>>>(h0, c0, c_pp[0], h_pp[0]);

    // 512 (row,m) tiles per step; 2 k-split waves per tile; 8 waves/block -> 128 blocks
    const int tiles_per_step = (B_SZ / 16) * (M_SZ / 16);   // 512
    for (int t = 0; t < T_STEPS; ++t) {
        const int pi = t & 1, po = pi ^ 1;                  // ping-pong: no intra-step races
        step_fused<<<tiles_per_step / 4, dim3(32, 8), 0, stream>>>(
            xcat, WallT, bias, q,
            h_pp[pi], c_pp[pi], h_pp[po], c_pp[po],
            (t == T_STEPS - 1) ? out : nullptr, t);
    }
}
